// ResGeneratorFixed_50843822850111
// MI455X (gfx1250) — compile-verified, matrix operand order checked by reading, not run on hardware
//
#include <hip/hip_runtime.h>
#include <hip/hip_bf16.h>
#include <math.h>

// ---------------------------------------------------------------------------
// Problem dims (from reference)
// ---------------------------------------------------------------------------
#define BATCH   1024
#define NOISE_D 256
#define RES_D   2048
#define NLAY    16
#define NC      90
#define NT      90
#define HEADS   (NC + NT)          // 180
#define OUTD    (NLAY * HEADS)     // 2880
#define WLEN    512
#define M_MAT   4

#define MOBILITY 0.0014
#define EPS_INF  3.9
#define EPS0     8.85e-12
#define M_STAR   (0.28 * 9.10938188e-31)
#define E_CH     1.60217646e-19
#define BN_EPS   1e-05f
#define PI_D     3.14159265358979323846

#define NBLK 4                     // N-tiles per wave (register blocking)

typedef __attribute__((ext_vector_type(16))) _Float16 v16h;
typedef __attribute__((ext_vector_type(8)))  float    v8f;

// ---------------------------------------------------------------------------
// f32 -> f16 conversion (weights / activations into WMMA-friendly storage)
// ---------------------------------------------------------------------------
__global__ void cvt_f32_f16(const float* __restrict__ in,
                            _Float16* __restrict__ out, int n) {
    int i = blockIdx.x * blockDim.x + threadIdx.x;
    if (i < n) out[i] = (_Float16)in[i];
}

// ---------------------------------------------------------------------------
// Register-blocked WMMA macro-tile: one wave computes a 16x(16*NBLK) strip of
// D = A[M,K] * W[N,K]^T  (f16 in, f32 acc). A-fragment is loaded once per
// K-step and reused across NBLK independent accumulators -> 4x less A traffic
// and 4 independent v_wmma per K-step (no D->A/B RAW between them).
//
// Fragment layouts per CDNA5 ISA 7.12.2 (16-bit A 16x32 / B 32x16):
//   lane 0..15  : row/col m=lane,    K = k0+[0..8) and k0+16+[0..8)
//   lane 16..31 : row/col m=lane-16, K = k0+8+[0..8) and k0+24+[0..8)
// Each lane issues contiguous 16B (b128) loads.
// ---------------------------------------------------------------------------
__device__ inline void wmma_strip_f16(const _Float16* __restrict__ A,
                                      const _Float16* __restrict__ W,
                                      int lda, int ldw, int K,
                                      int m0, int n0, int lane,
                                      v8f c[NBLK]) {
    const int hf = lane >> 4;        // K-half selector for A/B fragments
    const int mr = lane & 15;        // row (A) / col (B) within tile
    const _Float16* Ar = A + (size_t)(m0 + mr) * lda;
    const _Float16* Wr[NBLK];
#pragma unroll
    for (int t = 0; t < NBLK; ++t)
        Wr[t] = W + (size_t)(n0 + t * 16 + mr) * ldw;

    for (int k0 = 0; k0 < K; k0 += 32) {
        const int kk = k0 + hf * 8;
        v16h a;
#pragma unroll
        for (int i = 0; i < 8; ++i) {
            a[i]     = Ar[kk + i];
            a[8 + i] = Ar[kk + 16 + i];
        }
#pragma unroll
        for (int t = 0; t < NBLK; ++t) {
            v16h b;
#pragma unroll
            for (int i = 0; i < 8; ++i) {
                b[i]     = Wr[t][kk + i];
                b[8 + i] = Wr[t][kk + 16 + i];
            }
            // stream hint for next K-step of this weight row
            __builtin_prefetch(Wr[t] + kk + 32, 0, 1);
            c[t] = __builtin_amdgcn_wmma_f32_16x16x32_f16(
                       /*neg_a=*/false, a, /*neg_b=*/false, b,
                       /*c_mod=*/(short)0, c[t],
                       /*reuse_a=*/false, /*reuse_b=*/false);
        }
    }
}

// ---------------------------------------------------------------------------
// GEMM1: net = leaky_relu(noise @ W_init^T + b_init), stored as f16
// M=1024 N=2048 K=256; one wave per 16x64 strip (exact tiling => EXEC full)
// ---------------------------------------------------------------------------
__global__ void gemm1_wmma(const _Float16* __restrict__ A,   // [1024,256]
                           const _Float16* __restrict__ W,   // [2048,256]
                           const float* __restrict__ bias,   // [2048]
                           _Float16* __restrict__ net) {     // [1024,2048]
    const int wid  = blockIdx.x * (blockDim.x >> 5) + (threadIdx.x >> 5);
    const int lane = threadIdx.x & 31;
    const int ngrp = RES_D / (16 * NBLK);            // 32
    const int tm = wid / ngrp, tg = wid % ngrp;

    v8f c[NBLK] = {};
    wmma_strip_f16(A, W, NOISE_D, NOISE_D, NOISE_D,
                   tm * 16, tg * 16 * NBLK, lane, c);

    const int hf = lane >> 4, mr = lane & 15;
#pragma unroll
    for (int t = 0; t < NBLK; ++t) {
        const int col = tg * 16 * NBLK + t * 16 + mr;
        const float bv = bias[col];
#pragma unroll
        for (int r = 0; r < 8; ++r) {
            const int row = tm * 16 + r + 8 * hf;    // C layout: VGPR r -> M=r (+8 hi lanes)
            float x = c[t][r] + bv;
            x = (x > 0.0f) ? x : 0.2f * x;           // leaky_relu(0.2)
            net[(size_t)row * RES_D + col] = (_Float16)x;
        }
    }
}

// ---------------------------------------------------------------------------
// GEMM2: ct = BN(net @ W_end^T)   M=1024 N=2880 K=2048
// ---------------------------------------------------------------------------
__global__ void gemm2_wmma(const _Float16* __restrict__ A,   // [1024,2048]
                           const _Float16* __restrict__ W,   // [2880,2048]
                           const float* __restrict__ g,
                           const float* __restrict__ bta,
                           const float* __restrict__ mu,
                           const float* __restrict__ var,
                           float* __restrict__ ct) {         // [1024,2880]
    const int wid  = blockIdx.x * (blockDim.x >> 5) + (threadIdx.x >> 5);
    const int lane = threadIdx.x & 31;
    const int ngrp = OUTD / (16 * NBLK);             // 45
    const int tm = wid / ngrp, tg = wid % ngrp;

    v8f c[NBLK] = {};
    wmma_strip_f16(A, W, RES_D, RES_D, RES_D,
                   tm * 16, tg * 16 * NBLK, lane, c);

    const int hf = lane >> 4, mr = lane & 15;
#pragma unroll
    for (int t = 0; t < NBLK; ++t) {
        const int col = tg * 16 * NBLK + t * 16 + mr;
        const float s = g[col] * rsqrtf(var[col] + BN_EPS);
        const float o = bta[col] - mu[col] * s;
#pragma unroll
        for (int r = 0; r < 8; ++r) {
            const int row = tm * 16 + r + 8 * hf;
            ct[(size_t)row * OUTD + col] = c[t][r] * s + o;
        }
    }
}

// ---------------------------------------------------------------------------
// Dual softmax heads: one wave per (batch, layer). 90-wide softmax*weights,
// wave32 shuffle reductions. Writes wc (for drude), thicknesses and
// carrier_density outputs directly.
// ---------------------------------------------------------------------------
__device__ inline float wave_max(float v) {
#pragma unroll
    for (int m = 16; m >= 1; m >>= 1) v = fmaxf(v, __shfl_xor(v, m));
    return v;
}
__device__ inline float wave_sum(float v) {
#pragma unroll
    for (int m = 16; m >= 1; m >>= 1) v += __shfl_xor(v, m);
    return v;
}

__global__ void heads_kernel(const float* __restrict__ ct,
                             const int* __restrict__ alphaPtr,
                             const float* __restrict__ cr,   // [90]
                             const float* __restrict__ tr,   // [90]
                             float* __restrict__ wcbuf,      // [1024*16]
                             float* __restrict__ out_thick,  // [1024*16]
                             float* __restrict__ out_cd) {   // [1024*14]
    const int wid  = blockIdx.x * (blockDim.x >> 5) + (threadIdx.x >> 5);
    const int lane = threadIdx.x & 31;
    if (wid >= BATCH * NLAY) return;
    const int b = wid >> 4, l = wid & 15;
    const float alpha = (float)alphaPtr[0];
    const float* row = ct + (size_t)b * OUTD + l * HEADS;

    float res[2];
#pragma unroll
    for (int seg = 0; seg < 2; ++seg) {
        const float* base = row + seg * NC;
        const float* wv   = seg ? tr : cr;
        float m = -3.4e38f;
        for (int j = lane; j < NC; j += 32) m = fmaxf(m, base[j] * alpha);
        m = wave_max(m);
        float s = 0.0f, ws = 0.0f;
        for (int j = lane; j < NC; j += 32) {
            const float e = expf(base[j] * alpha - m);
            s += e; ws += e * wv[j];
        }
        s = wave_sum(s); ws = wave_sum(ws);
        res[seg] = ws / s;
    }
    if (lane == 0) {
        wcbuf[wid]     = res[0];
        out_thick[wid] = (l == 0 || l == NLAY - 1) ? __builtin_inff() : res[1];
        if (l >= 1 && l <= NLAY - 2) out_cd[b * (NLAY - 2) + (l - 1)] = res[0];
    }
}

// ---------------------------------------------------------------------------
// Drude model (f64, matches reference math) + boundary rows from n_database.
// Output interleaved complex128 (re, im), fully coalesced. This kernel writes
// the bandwidth-dominant 134 MB of output.
// ---------------------------------------------------------------------------
__global__ void drude_kernel(const float* __restrict__ wcbuf,  // [1024*16]
                             const double* __restrict__ kvec,  // [512]
                             const double* __restrict__ ndb,   // [4*512*2] interleaved
                             double* __restrict__ out) {       // [1024*16*512*2]
    const long long idx   = (long long)blockIdx.x * blockDim.x + threadIdx.x;
    const long long total = (long long)BATCH * NLAY * WLEN;
    if (idx >= total) return;
    const int w  = (int)(idx & (WLEN - 1));
    const int bl = (int)(idx >> 9);
    const int l  = bl & (NLAY - 1);

    double re, im;
    if (l == 0 || l == NLAY - 1) {
        const int rsel = (l == 0) ? 0 : (M_MAT - 1);
        re = ndb[((size_t)rsel * WLEN + w) * 2 + 0];
        im = ndb[((size_t)rsel * WLEN + w) * 2 + 1];
    } else {
        const double c      = (double)wcbuf[bl];
        const double lambda = (2.0 * PI_D / kvec[w]) * 1e-6;    // waves[w] um -> m
        const double omega  = 2.0 * PI_D * 3.0e8 / lambda;
        const double gamma  = E_CH / M_STAR / MOBILITY;
        const double nq     = c * 1e6;
        const double wp     = sqrt(nq * E_CH * E_CH / (EPS0 * M_STAR)) * 1e10;
        const double wp2    = wp * wp;
        const double den    = omega * omega + gamma * gamma;
        const double er     = EPS_INF - wp2 / den;
        const double ei     = wp2 * gamma / (omega * den);
        const double mag    = sqrt(er * er + ei * ei);
        re = sqrt(0.5 * er + 0.5 * mag);   // index_n
        im = sqrt(-0.5 * er + 0.5 * mag);  // index_k
    }
    out[idx * 2 + 0] = re;
    out[idx * 2 + 1] = im;
}

// ---------------------------------------------------------------------------
// Launcher
// ---------------------------------------------------------------------------
extern "C" void kernel_launch(void* const* d_in, const int* in_sizes, int n_in,
                              void* d_out, int out_size, void* d_ws, size_t ws_size,
                              hipStream_t stream) {
    const float*  noise   = (const float*)d_in[0];
    const int*    alpha   = (const int*)d_in[1];
    const float*  W_init  = (const float*)d_in[2];
    const float*  b_init  = (const float*)d_in[3];
    const float*  W_end   = (const float*)d_in[4];
    const float*  bn_g    = (const float*)d_in[5];
    const float*  bn_b    = (const float*)d_in[6];
    const float*  bn_m    = (const float*)d_in[7];
    const float*  bn_v    = (const float*)d_in[8];
    const double* n_db    = (const double*)d_in[9];   // complex128 interleaved
    const double* kvec    = (const double*)d_in[10];
    const float*  cr      = (const float*)d_in[11];
    const float*  tr      = (const float*)d_in[12];

    // workspace layout (bytes)
    char* ws = (char*)d_ws;
    size_t off = 0;
    _Float16* noise_h = (_Float16*)(ws + off); off += (size_t)BATCH * NOISE_D * 2;
    _Float16* winit_h = (_Float16*)(ws + off); off += (size_t)RES_D * NOISE_D * 2;
    _Float16* wend_h  = (_Float16*)(ws + off); off += (size_t)OUTD * RES_D * 2;
    _Float16* net_h   = (_Float16*)(ws + off); off += (size_t)BATCH * RES_D * 2;
    float*    ct      = (float*)   (ws + off); off += (size_t)BATCH * OUTD * 4;
    float*    wcbuf   = (float*)   (ws + off); off += (size_t)BATCH * NLAY * 4;

    // output layout: thicknesses f32 [1024*16] | drude complex128 [1024*16*512] | carrier f32 [1024*14]
    char* ob = (char*)d_out;
    float*  out_thick = (float*)ob;
    double* out_drude = (double*)(ob + (size_t)BATCH * NLAY * 4);
    float*  out_cd    = (float*)(ob + (size_t)BATCH * NLAY * 4 +
                                 (size_t)BATCH * NLAY * WLEN * 16);

    // 1) f32 -> f16 conversions
    {
        int n0 = BATCH * NOISE_D;
        cvt_f32_f16<<<(n0 + 255) / 256, 256, 0, stream>>>(noise, noise_h, n0);
        int n1 = RES_D * NOISE_D;
        cvt_f32_f16<<<(n1 + 255) / 256, 256, 0, stream>>>(W_init, winit_h, n1);
        int n2 = OUTD * RES_D;
        cvt_f32_f16<<<(n2 + 255) / 256, 256, 0, stream>>>(W_end, wend_h, n2);
    }

    // 2) GEMM1: 64 M-tiles x 32 N-groups = 2048 waves, 4 waves (128 thr)/block
    {
        int waves = (BATCH / 16) * (RES_D / (16 * NBLK));
        gemm1_wmma<<<waves / 4, 128, 0, stream>>>(noise_h, winit_h, b_init, net_h);
    }

    // 3) GEMM2: 64 M-tiles x 45 N-groups = 2880 waves
    {
        int waves = (BATCH / 16) * (OUTD / (16 * NBLK));
        gemm2_wmma<<<waves / 4, 128, 0, stream>>>(net_h, wend_h,
                                                  bn_g, bn_b, bn_m, bn_v, ct);
    }

    // 4) softmax heads: one wave per (b,l)
    {
        int waves = BATCH * NLAY;                            // 16384
        heads_kernel<<<waves / 4, 128, 0, stream>>>(ct, alpha, cr, tr,
                                                    wcbuf, out_thick, out_cd);
    }

    // 5) drude + boundary rows
    {
        long long total = (long long)BATCH * NLAY * WLEN;    // 8.4M
        drude_kernel<<<(unsigned)((total + 255) / 256), 256, 0, stream>>>(
            wcbuf, kvec, n_db, out_drude);
    }
}
